// SimpleGAT_5669356831046
// MI455X (gfx1250) — matrix operand order repaired and checked by reading, not compiled, not run on hardware
//
#include <hip/hip_runtime.h>
#include <hip/hip_bf16.h>
#include <math.h>

typedef _Float16 half_t;
typedef __attribute__((ext_vector_type(16))) _Float16 v16h;
typedef __attribute__((ext_vector_type(2)))  _Float16 v2h;
typedef __attribute__((ext_vector_type(8)))  float    v8f;

#define NNODES 16384
#define NGRAPH 256
#define NPG    64      // nodes per graph (contiguous)
#define NHEAD  8
#define FDIM   64
#define DOUT   512     // NHEAD*FDIM

// ---------------------------------------------------------------- utilities

__global__ void fill_f32(float* __restrict__ p, long n, float v) {
  long i = (long)blockIdx.x * blockDim.x + threadIdx.x;
  long stride = (long)gridDim.x * blockDim.x;
  for (; i < n; i += stride) p[i] = v;
}

__global__ void f32_to_f16(const float* __restrict__ src, half_t* __restrict__ dst, long n) {
  long i = (long)blockIdx.x * blockDim.x + threadIdx.x;
  long stride = (long)gridDim.x * blockDim.x;
  for (; i < n; i += stride) dst[i] = (half_t)src[i];
}

// Pack weight matrix B[K x Nc] (f32, row-major) into WMMA-B fragment-major f16:
//   Bp[(((c*(K/32) + t)*32 + lane)*16 + j] = B[(t*32 + (lane>>4)*16 + j)*Nc + c*16 + (lane&15)]
// so a wave's B fragment for (col-tile c, k-tile t) is 32 contiguous v16h's,
// one aligned 32-byte load per lane (2 x global_load_b128).
__global__ void pack_b_f16(const float* __restrict__ B, half_t* __restrict__ Bp,
                           int K, int Nc) {
  long total = (long)K * Nc;
  long i = (long)blockIdx.x * blockDim.x + threadIdx.x;
  long stride = (long)gridDim.x * blockDim.x;
  int kt = K >> 5;                       // k-tiles
  for (; i < total; i += stride) {
    int  j    = (int)(i & 15);
    long frag = i >> 4;
    int  lane = (int)(frag & 31);
    long ct   = frag >> 5;
    int  t    = (int)(ct % kt);
    int  c    = (int)(ct / kt);
    int  krow = t * 32 + (lane >> 4) * 16 + j;
    int  col  = c * 16 + (lane & 15);
    Bp[i] = (half_t)B[(long)krow * Nc + col];
  }
}

// ------------------------------------------------------- WMMA GEMM (f16->f32)
// C[M x Nc] = A[M x K] * B[K x Nc]. A row-major f16, B pre-packed fragment-major.
// Block = 128 threads = 4 waves; wave w owns rows [blockIdx.x*64 + w*16, +16) and
// a 64-wide column strip (4 accumulators, 4 v_wmma per k-step sharing one A frag).
__global__ __launch_bounds__(128) void wmma_gemm(
    const half_t* __restrict__ A, const half_t* __restrict__ Bp,
    float* __restrict__ C, int M, int K, int Nc) {
  const int lane = threadIdx.x & 31;
  const int wave = threadIdx.x >> 5;
  const int row0 = (blockIdx.x * 4 + wave) * 16;
  const int ct0  = blockIdx.y * 4;           // first of 4 column tiles
  const int r    = lane & 15;
  const int hi   = lane >> 4;                // 0: lanes 0-15, 1: lanes 16-31
  const long arow = row0 + r;
  const int kt   = K >> 5;                   // k-tiles of 32
  const v16h* __restrict__ Bf = (const v16h*)Bp;

  v8f c0 = {}, c1 = {}, c2 = {}, c3 = {};
  const half_t* Ar = A + arow * K;

  for (int t = 0; t < kt; ++t) {
    const int k0 = t * 32;
    // prefetch next A k-tile while this step's WMMAs run (global_prefetch_b8)
    __builtin_prefetch(Ar + k0 + 32, 0, 1);
    // A fragment: one contiguous 16-half chunk per lane (k = 16*hi + s)
    v16h a;
    #pragma unroll
    for (int v = 0; v < 8; ++v) {
      int kb = 2 * v + hi * 16;
      v2h p = *(const v2h*)(Ar + k0 + kb);
      a[2 * v]     = p[0];
      a[2 * v + 1] = p[1];
    }
    // B fragments: one aligned v16h load each (2 x b128), then WMMA
    v16h b0 = Bf[((long)(ct0 + 0) * kt + t) * 32 + lane];
    v16h b1 = Bf[((long)(ct0 + 1) * kt + t) * 32 + lane];
    v16h b2 = Bf[((long)(ct0 + 2) * kt + t) * 32 + lane];
    v16h b3 = Bf[((long)(ct0 + 3) * kt + t) * 32 + lane];
    c0 = __builtin_amdgcn_wmma_f32_16x16x32_f16(false, a, false, b0, (short)0, c0, false, false);
    c1 = __builtin_amdgcn_wmma_f32_16x16x32_f16(false, a, false, b1, (short)0, c1, false, false);
    c2 = __builtin_amdgcn_wmma_f32_16x16x32_f16(false, a, false, b2, (short)0, c2, false, false);
    c3 = __builtin_amdgcn_wmma_f32_16x16x32_f16(false, a, false, b3, (short)0, c3, false, false);
  }

  // C/D layout: VGPR v -> M = v + 8*hi, N = lane&15
  #pragma unroll
  for (int v = 0; v < 8; ++v) {
    long rr = (long)(row0 + v + hi * 8) * Nc + ct0 * 16 + r;
    C[rr]            = c0[v];
    C[rr + 16]       = c1[v];
    C[rr + 32]       = c2[v];
    C[rr + 48]       = c3[v];
  }
}

// ----------------------------------------------------------- edge kernels

__device__ inline void atomicMaxF(float* addr, float val) {
  if (val >= 0.0f) atomicMax((int*)addr, __float_as_int(val));
  else             atomicMin((unsigned int*)addr, __float_as_uint(val));
}

// one thread per (edge, head): logit[e,h] = sum_f leaky(xl[src]+xr[dst]) * att
__global__ void edge_logit(const float* __restrict__ xl, const float* __restrict__ xr,
                           const float* __restrict__ att,
                           const int* __restrict__ src, const int* __restrict__ dst,
                           float* __restrict__ logit, int E) {
  long i = (long)blockIdx.x * blockDim.x + threadIdx.x;
  if (i >= (long)E * NHEAD) return;
  int e = (int)(i >> 3), h = (int)(i & 7);
  long sb = (long)src[e] * DOUT + h * FDIM;
  long db = (long)dst[e] * DOUT + h * FDIM;
  const float* ah = att + h * FDIM;
  float acc = 0.0f;
  for (int f = 0; f < FDIM; ++f) {
    float m = xl[sb + f] + xr[db + f];
    m = (m > 0.0f) ? m : 0.2f * m;
    acc += m * ah[f];
  }
  logit[i] = acc;
}

__global__ void edge_max(const float* __restrict__ logit, const int* __restrict__ dst,
                         float* __restrict__ lmax, int E) {
  long i = (long)blockIdx.x * blockDim.x + threadIdx.x;
  if (i >= (long)E * NHEAD) return;
  int e = (int)(i >> 3), h = (int)(i & 7);
  atomicMaxF(&lmax[(long)dst[e] * NHEAD + h], logit[i]);
}

__global__ void edge_expsum(float* __restrict__ logit, const int* __restrict__ dst,
                            const float* __restrict__ lmax, float* __restrict__ den, int E) {
  long i = (long)blockIdx.x * blockDim.x + threadIdx.x;
  if (i >= (long)E * NHEAD) return;
  int e = (int)(i >> 3), h = (int)(i & 7);
  long d = (long)dst[e] * NHEAD + h;
  float lm = lmax[d];
  if (lm < -1e37f) lm = 0.0f;            // -inf guard per reference
  float p = expf(logit[i] - lm);
  logit[i] = p;                           // overwrite with p
  atomicAdd(&den[d], p);
}

// one thread per (edge, dim): agg[dst, d] += alpha * xl[src, d]
__global__ void edge_agg(const float* __restrict__ p, const float* __restrict__ den,
                         const float* __restrict__ xl,
                         const int* __restrict__ src, const int* __restrict__ dst,
                         float* __restrict__ agg, long total) {
  long i = (long)blockIdx.x * blockDim.x + threadIdx.x;
  long stride = (long)gridDim.x * blockDim.x;
  for (; i < total; i += stride) {
    int e = (int)(i >> 9);                // / DOUT
    int d = (int)(i & 511);
    int h = d >> 6;
    int dn = dst[e];
    float alpha = p[(long)e * NHEAD + h] / (den[(long)dn * NHEAD + h] + 1e-16f);
    atomicAdd(&agg[(long)dn * DOUT + d], alpha * xl[(long)src[e] * DOUT + d]);
  }
}

// -------------------------------------------- GraphNorm (+bias) + ReLU, in place
// batch = repeat(arange(G), 64): graph g owns nodes [g*64, g*64+64).
__global__ void graphnorm_relu(float* __restrict__ x, const float* __restrict__ bias,
                               const float* __restrict__ w, const float* __restrict__ b,
                               const float* __restrict__ ms, int D) {
  int g = blockIdx.x;
  long base = (long)g * NPG;
  for (int d = threadIdx.x; d < D; d += blockDim.x) {
    float bd = bias ? bias[d] : 0.0f;
    float mean = 0.0f;
    for (int n = 0; n < NPG; ++n) mean += x[(base + n) * D + d] + bd;
    mean *= (1.0f / NPG);
    float m = ms[d] * mean;
    float var = 0.0f;
    for (int n = 0; n < NPG; ++n) {
      float xc = x[(base + n) * D + d] + bd - m;
      var += xc * xc;
    }
    var *= (1.0f / NPG);
    float inv = rsqrtf(var + 1e-5f) * w[d];
    float bb = b[d];
    for (int n = 0; n < NPG; ++n) {
      float xc = x[(base + n) * D + d] + bd - m;
      float y = xc * inv + bb;
      x[(base + n) * D + d] = fmaxf(y, 0.0f);
    }
  }
}

// layer3: concat=False -> mean over heads, then + b3
__global__ void headmean(const float* __restrict__ agg, const float* __restrict__ b3,
                         float* __restrict__ h3, long total) {
  long i = (long)blockIdx.x * blockDim.x + threadIdx.x;
  if (i >= total) return;
  long n = i >> 6;
  int f = (int)(i & 63);
  float s = 0.0f;
  for (int h = 0; h < NHEAD; ++h) s += agg[n * DOUT + h * FDIM + f];
  h3[i] = s * 0.125f + b3[f];
}

// global_mean_pool + final linear (64 -> 2)
__global__ void pool_linear(const float* __restrict__ h3, const float* __restrict__ Wlin,
                            const float* __restrict__ blin, float* __restrict__ out) {
  __shared__ float s[FDIM];
  int g = blockIdx.x, f = threadIdx.x;
  long base = (long)g * NPG * FDIM;
  float acc = 0.0f;
  for (int n = 0; n < NPG; ++n) acc += h3[base + (long)n * FDIM + f];
  s[f] = acc * (1.0f / NPG);
  __syncthreads();
  if (f < 2) {
    float o = blin[f];
    for (int k = 0; k < FDIM; ++k) o += s[k] * Wlin[k * 2 + f];
    out[g * 2 + f] = o;
  }
}

// ---------------------------------------------------------------- launcher

extern "C" void kernel_launch(void* const* d_in, const int* in_sizes, int n_in,
                              void* d_out, int out_size, void* d_ws, size_t ws_size,
                              hipStream_t stream) {
  const float* x    = (const float*)d_in[0];
  const int*   esrc = (const int*)d_in[1];
  const int*   edst = (const int*)d_in[2];
  const int E = in_sizes[1];                       // 147456 (with self loops)

  const float* Wl[3]  = {(const float*)d_in[4],  (const float*)d_in[11], (const float*)d_in[18]};
  const float* Wr[3]  = {(const float*)d_in[5],  (const float*)d_in[12], (const float*)d_in[19]};
  const float* att[3] = {(const float*)d_in[6],  (const float*)d_in[13], (const float*)d_in[20]};
  const float* bia[3] = {(const float*)d_in[7],  (const float*)d_in[14], (const float*)d_in[21]};
  const float* gnw[3] = {(const float*)d_in[8],  (const float*)d_in[15], (const float*)d_in[22]};
  const float* gnb[3] = {(const float*)d_in[9],  (const float*)d_in[16], (const float*)d_in[23]};
  const float* gnm[3] = {(const float*)d_in[10], (const float*)d_in[17], (const float*)d_in[24]};
  const float* Wlin = (const float*)d_in[25];
  const float* blin = (const float*)d_in[26];
  float* out = (float*)d_out;

  const int Din[3] = {256, 512, 512};

  // ---- carve workspace (~155 MB)
  char* w = (char*)d_ws;
  auto alloc = [&](size_t bytes) -> void* {
    void* p = (void*)w;
    w += (bytes + 255) & ~(size_t)255;
    return p;
  };
  float*  buf0  = (float*)alloc((size_t)NNODES * DOUT * 4);
  float*  buf1  = (float*)alloc((size_t)NNODES * DOUT * 4);
  float*  xl    = (float*)alloc((size_t)NNODES * DOUT * 4);
  float*  xr    = (float*)alloc((size_t)NNODES * DOUT * 4);
  half_t* x16   = (half_t*)alloc((size_t)NNODES * DOUT * 2);
  half_t* wl16[3]; half_t* wr16[3];
  for (int i = 0; i < 3; ++i) {
    wl16[i] = (half_t*)alloc((size_t)Din[i] * DOUT * 2);
    wr16[i] = (half_t*)alloc((size_t)Din[i] * DOUT * 2);
  }
  float* logit = (float*)alloc((size_t)E * NHEAD * 4);
  float* lmaxb = (float*)alloc((size_t)NNODES * NHEAD * 4);
  float* denb  = (float*)alloc((size_t)NNODES * NHEAD * 4);
  float* h3    = (float*)alloc((size_t)NNODES * FDIM * 4);

  // ---- convert + pack weights into WMMA-B fragment layout (f16) once
  for (int i = 0; i < 3; ++i) {
    pack_b_f16<<<1024, 256, 0, stream>>>(Wl[i], wl16[i], Din[i], DOUT);
    pack_b_f16<<<1024, 256, 0, stream>>>(Wr[i], wr16[i], Din[i], DOUT);
  }

  const long eh = (long)E * NHEAD;
  const int  ehBlocks = (int)((eh + 255) / 256);
  float* bufs[2] = {buf0, buf1};

  for (int layer = 0; layer < 3; ++layer) {
    const float* cur = (layer == 0) ? x : bufs[(layer + 1) & 1];
    float* agg = bufs[layer & 1];
    int K = Din[layer];

    // f16 staging of layer input
    f32_to_f16<<<2048, 256, 0, stream>>>(cur, x16, (long)NNODES * K);

    // xl = cur @ Wl, xr = cur @ Wr  (WMMA; 4 waves x 4 col-tiles per block)
    dim3 ggrid(NNODES / 64, DOUT / 64);
    wmma_gemm<<<ggrid, 128, 0, stream>>>(x16, wl16[layer], xl, NNODES, K, DOUT);
    wmma_gemm<<<ggrid, 128, 0, stream>>>(x16, wr16[layer], xr, NNODES, K, DOUT);

    // init segment buffers
    fill_f32<<<512, 256, 0, stream>>>(lmaxb, (long)NNODES * NHEAD, -INFINITY);
    fill_f32<<<512, 256, 0, stream>>>(denb,  (long)NNODES * NHEAD, 0.0f);
    fill_f32<<<4096, 256, 0, stream>>>(agg,  (long)NNODES * DOUT,  0.0f);

    // attention: logits, segment-max, exp/sum, weighted aggregate
    edge_logit<<<ehBlocks, 256, 0, stream>>>(xl, xr, att[layer], esrc, edst, logit, E);
    edge_max<<<ehBlocks, 256, 0, stream>>>(logit, edst, lmaxb, E);
    edge_expsum<<<ehBlocks, 256, 0, stream>>>(logit, edst, lmaxb, denb, E);
    edge_agg<<<65536, 256, 0, stream>>>(logit, denb, xl, esrc, edst, agg, (long)E * DOUT);

    if (layer < 2) {
      // + bias, GraphNorm, ReLU (in place)
      graphnorm_relu<<<NGRAPH, 256, 0, stream>>>(agg, bia[layer], gnw[layer], gnb[layer],
                                                 gnm[layer], DOUT);
    } else {
      // mean over heads + b3, GraphNorm(D=64), ReLU, pool + linear
      long t = (long)NNODES * FDIM;
      headmean<<<(int)((t + 255) / 256), 256, 0, stream>>>(agg, bia[2], h3, t);
      graphnorm_relu<<<NGRAPH, 256, 0, stream>>>(h3, nullptr, gnw[2], gnb[2], gnm[2], FDIM);
      pool_linear<<<NGRAPH, FDIM, 0, stream>>>(h3, Wlin, blin, out);
    }
  }
}
